// ProteinGAT_69630009803000
// MI455X (gfx1250) — compile-verified, matrix operand order checked by reading, not run on hardware
//
#include <hip/hip_runtime.h>
#include <hip/hip_bf16.h>

// ---------------------------------------------------------------------------
// Types for CDNA5 WMMA (wave32, 16x16x32 bf16 -> f32)
// ---------------------------------------------------------------------------
typedef __bf16 bf16_t;
typedef __attribute__((ext_vector_type(16))) __bf16 v16bf;
typedef __attribute__((ext_vector_type(8)))  float  v8f;
typedef __attribute__((ext_vector_type(4)))  unsigned int u32x4;

union FragBF { v16bf v; u32x4 q[2]; };

#define NN   50000
#define EE   800000
#define ETOT 850000     // EE + NN self loops
#define IND  329
#define KP1  352        // 329 padded to 11*32
#define C1   256        // heads*hid conv1
#define HID  64

__device__ inline bf16_t f2bf(float f) {
    unsigned u = __float_as_uint(f);
    u += 0x7FFFu + ((u >> 16) & 1u);            // round to nearest even
    unsigned short s = (unsigned short)(u >> 16);
    return __builtin_bit_cast(bf16_t, s);
}

// monotonic float<->uint mapping for atomic max on floats
__device__ inline unsigned fmap(float f) {
    unsigned u = __float_as_uint(f);
    return (u & 0x80000000u) ? ~u : (u | 0x80000000u);
}
__device__ inline float funmap(unsigned u) {
    unsigned v = (u & 0x80000000u) ? (u & 0x7FFFFFFFu) : ~u;
    return __uint_as_float(v);
}

__device__ inline v8f wmma_bf16(v16bf a, v16bf b, v8f c) {
    return __builtin_amdgcn_wmma_f32_16x16x32_bf16(
        /*neg_a=*/false, a, /*neg_b=*/false, b,
        /*c_mod=*/(short)0, c, /*reuse_a=*/false, /*reuse_b=*/false);
}

// CDNA5 async copy: global -> LDS, 16 bytes, tracked by ASYNCcnt.
// ldsoff = byte offset inside workgroup LDS allocation; g = 64-bit global addr.
__device__ inline void async_g2l_b128(unsigned ldsoff, const void* g) {
    asm volatile("global_load_async_to_lds_b128 %0, %1, off"
                 :: "v"(ldsoff), "v"(g) : "memory");
}
__device__ inline void wait_async0() {
    asm volatile("s_wait_asynccnt 0x0" ::: "memory");
}

// ---------------------------------------------------------------------------
// Prep kernels
// ---------------------------------------------------------------------------

// sum of edge_attr over E edges (two components) -> scal[0..1]
__global__ void k_sum_ea(const float* __restrict__ ea, float* __restrict__ scal) {
    int i = blockIdx.x * blockDim.x + threadIdx.x;
    float v0 = 0.f, v1 = 0.f;
    if (i < EE) { v0 = ea[2 * i]; v1 = ea[2 * i + 1]; }
    for (int o = 16; o; o >>= 1) { v0 += __shfl_down(v0, o); v1 += __shfl_down(v1, o); }
    if ((threadIdx.x & 31) == 0) { atomicAdd(&scal[0], v0); atomicAdd(&scal[1], v1); }
}

// scal layout: [0..1]=sum_ea, [2..3]=mean_ea, [4..11]=q1[d][h], [12..13]=q2[d]
__global__ void k_prep(const float* __restrict__ We1, const float* __restrict__ atte1,
                       const float* __restrict__ We2, const float* __restrict__ atte2,
                       float* __restrict__ scal) {
    int t = threadIdx.x;
    if (t < 8) {                       // q1[d][h] = sum_c We1[d, h*64+c] * att_e1[h, c]
        int d = t >> 2, h = t & 3;
        float s = 0.f;
        for (int c = 0; c < 64; ++c)
            s += We1[d * C1 + h * 64 + c] * atte1[h * 64 + c];
        scal[4 + t] = s;
    } else if (t < 10) {               // q2[d] = sum_c We2[d, c] * att_e2[c]
        int d = t - 8;
        float s = 0.f;
        for (int c = 0; c < 64; ++c)
            s += We2[d * HID + c] * atte2[c];
        scal[12 + d] = s;
    } else if (t < 12) {               // mean_ea
        int d = t - 10;
        scal[2 + d] = scal[d] / (float)EE;
    }
}

// x (N,329) f32 -> xb (N,352) bf16, zero padded
__global__ void k_cvt_x(const float* __restrict__ x, bf16_t* __restrict__ xb) {
    int gid = blockIdx.x * blockDim.x + threadIdx.x;
    if (gid >= NN * KP1) return;
    int n = gid / KP1, k = gid - n * KP1;
    xb[gid] = (k < IND) ? f2bf(x[(size_t)n * IND + k]) : f2bf(0.f);
}

// W1 (329,256) -> w1bt (256,352) bf16 transposed, zero padded
__global__ void k_cvt_w1t(const float* __restrict__ W1, bf16_t* __restrict__ w1bt) {
    int gid = blockIdx.x * blockDim.x + threadIdx.x;
    if (gid >= C1 * KP1) return;
    int n = gid / KP1, k = gid - n * KP1;
    w1bt[gid] = (k < IND) ? f2bf(W1[(size_t)k * C1 + n]) : f2bf(0.f);
}

// W2 (256,64) -> w2t (64,256) bf16 transposed
__global__ void k_cvt_w2t(const float* __restrict__ W2, bf16_t* __restrict__ w2t) {
    int gid = blockIdx.x * blockDim.x + threadIdx.x;
    if (gid >= HID * C1) return;
    int n = gid >> 8, k = gid & 255;
    w2t[gid] = f2bf(W2[(size_t)k * HID + n]);
}

// ---------------------------------------------------------------------------
// GEMM1: h1[N,256] = xb[N,352] * W1  (Bt = w1bt[256,352] bf16)
// One block = 4 waves = one 16-row M tile x all 256 cols.
// B panel k-chunks (256 rows x 32 k = 16 KB) are double-buffered in LDS and
// prefetched with global_load_async_to_lds_b128 (ASYNCcnt) while WMMAs run.
// ---------------------------------------------------------------------------
__global__ __launch_bounds__(128) void k_gemm1(const bf16_t* __restrict__ A,
                                               const bf16_t* __restrict__ Bt,
                                               float* __restrict__ C) {
    __shared__ __align__(16) unsigned char bsm[2][16384];   // 2 x 256 rows x 64B

    const int tid  = threadIdx.x;
    const int lane = tid & 31;
    const int wv   = tid >> 5;
    const int m0   = blockIdx.x * 16;
    const int n0   = wv * 64;
    const int r    = lane & 15;
    const int hi   = lane >> 4;
    const int koff = hi << 3;

    const unsigned lb0 = (unsigned)(uintptr_t)&bsm[0][0];
    const unsigned lb1 = (unsigned)(uintptr_t)&bsm[1][0];

    // stage k-chunk [kb, kb+32) of all 256 B rows into buffer `buf`
    auto stage = [&](int kb, int buf) {
        unsigned lb = buf ? lb1 : lb0;
#pragma unroll
        for (int i = 0; i < 8; ++i) {
            int e   = tid * 8 + i;          // 1024 x 16B vectors
            int row = e >> 2;
            int seg = e & 3;
            const char* g = (const char*)Bt + ((size_t)row * KP1 + kb) * 2 + seg * 16;
            async_g2l_b128(lb + row * 64 + seg * 16, g);
        }
    };

    const bf16_t* arow = A + (size_t)(m0 + r) * KP1 + koff;

    // prologue: stage chunk 0
    stage(0, 0);
    wait_async0();
    __syncthreads();

    v8f acc0 = {}, acc1 = {}, acc2 = {}, acc3 = {};
    for (int it = 0; it < KP1 / 32; ++it) {
        int buf = it & 1;
        if (it + 1 < KP1 / 32) stage((it + 1) * 32, buf ^ 1);   // overlap with compute

        FragBF a;
        a.q[0] = *(const u32x4*)(arow + it * 32);
        a.q[1] = *(const u32x4*)(arow + it * 32 + 16);

        const unsigned char* bs = bsm[buf];
        FragBF b0, b1, b2, b3;
        {
            const unsigned char* p = bs + (n0 + 0 + r) * 64 + koff * 2;
            b0.q[0] = *(const u32x4*)p; b0.q[1] = *(const u32x4*)(p + 32);
        }
        {
            const unsigned char* p = bs + (n0 + 16 + r) * 64 + koff * 2;
            b1.q[0] = *(const u32x4*)p; b1.q[1] = *(const u32x4*)(p + 32);
        }
        {
            const unsigned char* p = bs + (n0 + 32 + r) * 64 + koff * 2;
            b2.q[0] = *(const u32x4*)p; b2.q[1] = *(const u32x4*)(p + 32);
        }
        {
            const unsigned char* p = bs + (n0 + 48 + r) * 64 + koff * 2;
            b3.q[0] = *(const u32x4*)p; b3.q[1] = *(const u32x4*)(p + 32);
        }
        acc0 = wmma_bf16(a.v, b0.v, acc0);
        acc1 = wmma_bf16(a.v, b1.v, acc1);
        acc2 = wmma_bf16(a.v, b2.v, acc2);
        acc3 = wmma_bf16(a.v, b3.v, acc3);

        wait_async0();      // staged chunk complete (this wave's share)
        __syncthreads();    // all waves' shares visible; also protects overwrite
    }

    const int col = lane & 15;
    const int ro  = hi << 3;
#pragma unroll
    for (int i = 0; i < 8; ++i) {
        size_t base = (size_t)(m0 + ro + i) * C1 + n0 + col;
        C[base]      = acc0[i];
        C[base + 16] = acc1[i];
        C[base + 32] = acc2[i];
        C[base + 48] = acc3[i];
    }
}

// ---------------------------------------------------------------------------
// GEMM2: h2[N,64] = h1b[N,256] * W2  (Bt = w2t[64,256] bf16, 32 KB total).
// Whole B is async-staged into LDS once; 4 waves/block, one M tile each.
// ---------------------------------------------------------------------------
__global__ __launch_bounds__(128) void k_gemm2(const bf16_t* __restrict__ A,
                                               const bf16_t* __restrict__ Bt,
                                               float* __restrict__ C) {
    __shared__ __align__(16) unsigned char bsm[32768];      // 64 rows x 512B

    const int tid  = threadIdx.x;
    const int lane = tid & 31;
    const int wv   = tid >> 5;
    const int mt   = blockIdx.x * 4 + wv;
    const int r    = lane & 15;
    const int hi   = lane >> 4;
    const int koff = hi << 3;

    const unsigned lb = (unsigned)(uintptr_t)&bsm[0];
#pragma unroll
    for (int i = 0; i < 16; ++i) {                           // 2048 x 16B vectors
        int e = tid * 16 + i;
        async_g2l_b128(lb + e * 16, (const char*)Bt + (size_t)e * 16);
    }
    wait_async0();
    __syncthreads();

    if (mt < NN / 16) {
        const int m0 = mt * 16;
        const bf16_t* arow = A + (size_t)(m0 + r) * C1 + koff;

        v8f acc0 = {}, acc1 = {}, acc2 = {}, acc3 = {};
        for (int kb = 0; kb < C1; kb += 32) {
            FragBF a;
            a.q[0] = *(const u32x4*)(arow + kb);
            a.q[1] = *(const u32x4*)(arow + kb + 16);

            FragBF b0, b1, b2, b3;
            {
                const unsigned char* p = bsm + ( 0 + r) * 512 + (kb + koff) * 2;
                b0.q[0] = *(const u32x4*)p; b0.q[1] = *(const u32x4*)(p + 32);
            }
            {
                const unsigned char* p = bsm + (16 + r) * 512 + (kb + koff) * 2;
                b1.q[0] = *(const u32x4*)p; b1.q[1] = *(const u32x4*)(p + 32);
            }
            {
                const unsigned char* p = bsm + (32 + r) * 512 + (kb + koff) * 2;
                b2.q[0] = *(const u32x4*)p; b2.q[1] = *(const u32x4*)(p + 32);
            }
            {
                const unsigned char* p = bsm + (48 + r) * 512 + (kb + koff) * 2;
                b3.q[0] = *(const u32x4*)p; b3.q[1] = *(const u32x4*)(p + 32);
            }
            acc0 = wmma_bf16(a.v, b0.v, acc0);
            acc1 = wmma_bf16(a.v, b1.v, acc1);
            acc2 = wmma_bf16(a.v, b2.v, acc2);
            acc3 = wmma_bf16(a.v, b3.v, acc3);
        }
        const int col = lane & 15;
        const int ro  = hi << 3;
#pragma unroll
        for (int i = 0; i < 8; ++i) {
            size_t base = (size_t)(m0 + ro + i) * HID + col;
            C[base]      = acc0[i];
            C[base + 16] = acc1[i];
            C[base + 32] = acc2[i];
            C[base + 48] = acc3[i];
        }
    }
}

// ---------------------------------------------------------------------------
// Node attention scores, conv1: one wave per node, 8 lanes per head
// ---------------------------------------------------------------------------
__global__ void k_att1(const float* __restrict__ h1, const float* __restrict__ asrc,
                       const float* __restrict__ adst, float* __restrict__ as1,
                       float* __restrict__ ad1) {
    int wid = (blockIdx.x * blockDim.x + threadIdx.x) >> 5;
    if (wid >= NN) return;
    int lane = threadIdx.x & 31;
    const float* hr = h1   + (size_t)wid * C1 + lane * 8;
    const float* sr = asrc + lane * 8;
    const float* dr = adst + lane * 8;
    float ps = 0.f, pd = 0.f;
#pragma unroll
    for (int i = 0; i < 8; ++i) { float hv = hr[i]; ps += hv * sr[i]; pd += hv * dr[i]; }
    ps += __shfl_xor(ps, 1); ps += __shfl_xor(ps, 2); ps += __shfl_xor(ps, 4);
    pd += __shfl_xor(pd, 1); pd += __shfl_xor(pd, 2); pd += __shfl_xor(pd, 4);
    if ((lane & 7) == 0) {
        int h = lane >> 3;
        as1[wid * 4 + h] = ps;
        ad1[wid * 4 + h] = pd;
    }
}

// conv2: one wave per node, full-wave reduction
__global__ void k_att2(const float* __restrict__ h2, const float* __restrict__ asrc,
                       const float* __restrict__ adst, float* __restrict__ as2,
                       float* __restrict__ ad2) {
    int wid = (blockIdx.x * blockDim.x + threadIdx.x) >> 5;
    if (wid >= NN) return;
    int lane = threadIdx.x & 31;
    const float* hr = h2 + (size_t)wid * HID;
    float ps = hr[lane] * asrc[lane] + hr[lane + 32] * asrc[lane + 32];
    float pd = hr[lane] * adst[lane] + hr[lane + 32] * adst[lane + 32];
    for (int o = 16; o; o >>= 1) { ps += __shfl_xor(ps, o); pd += __shfl_xor(pd, o); }
    if (lane == 0) { as2[wid] = ps; ad2[wid] = pd; }
}

// ---------------------------------------------------------------------------
// Edge softmax, conv1 (H=4)
// ---------------------------------------------------------------------------
__global__ void k_alpha1(const int* __restrict__ ei, const float* __restrict__ ea,
                         const float* __restrict__ scal, const float* __restrict__ as1,
                         const float* __restrict__ ad1, float* __restrict__ attn,
                         unsigned* __restrict__ amax) {
    int et = blockIdx.x * blockDim.x + threadIdx.x;
    if (et >= ETOT) return;
    int s, d; float e0, e1;
    if (et < EE) { s = ei[et]; d = ei[EE + et]; e0 = ea[2 * et]; e1 = ea[2 * et + 1]; }
    else         { s = d = et - EE;             e0 = scal[2];    e1 = scal[3]; }
#pragma unroll
    for (int h = 0; h < 4; ++h) {
        float a = as1[s * 4 + h] + ad1[d * 4 + h] + e0 * scal[4 + h] + e1 * scal[8 + h];
        a = a > 0.f ? a : 0.2f * a;
        attn[(size_t)et * 4 + h] = a;
        atomicMax(&amax[d * 4 + h], fmap(a));
    }
}

__global__ void k_ex1(const int* __restrict__ ei, float* __restrict__ attn,
                      const unsigned* __restrict__ amax, float* __restrict__ den) {
    int et = blockIdx.x * blockDim.x + threadIdx.x;
    if (et >= ETOT) return;
    int d = (et < EE) ? ei[EE + et] : et - EE;
#pragma unroll
    for (int h = 0; h < 4; ++h) {
        float a = attn[(size_t)et * 4 + h];
        float e = __expf(a - funmap(amax[d * 4 + h]));
        attn[(size_t)et * 4 + h] = e;
        atomicAdd(&den[d * 4 + h], e);
    }
}

// aggregation conv1: one wave per edge, 8 channels per lane, atomic scatter
__global__ void k_agg1(const int* __restrict__ ei, const float* __restrict__ attn,
                       const float* __restrict__ den, const float* __restrict__ h1,
                       float* __restrict__ out1) {
    int eid = (blockIdx.x * blockDim.x + threadIdx.x) >> 5;
    if (eid >= ETOT) return;
    int lane = threadIdx.x & 31;
    int s, d;
    if (eid < EE) { s = ei[eid]; d = ei[EE + eid]; } else { s = d = eid - EE; }
    int h = lane >> 3;
    float w = attn[(size_t)eid * 4 + h] / (den[d * 4 + h] + 1e-16f);
    const float* hr = h1 + (size_t)s * C1 + lane * 8;
    float* orow = out1 + (size_t)d * C1 + lane * 8;
#pragma unroll
    for (int i = 0; i < 8; ++i) atomicAdd(&orow[i], w * hr[i]);
}

// bias + ELU, then convert to bf16 for GEMM2
__global__ void k_elu(const float* __restrict__ out1, const float* __restrict__ b1,
                      bf16_t* __restrict__ h1b) {
    int gid = blockIdx.x * blockDim.x + threadIdx.x;
    if (gid >= NN * C1) return;
    float v = out1[gid] + b1[gid & 255];
    v = v > 0.f ? v : (__expf(v) - 1.f);
    h1b[gid] = f2bf(v);
}

// ---------------------------------------------------------------------------
// Edge softmax, conv2 (H=1)
// ---------------------------------------------------------------------------
__global__ void k_alpha2(const int* __restrict__ ei, const float* __restrict__ ea,
                         const float* __restrict__ scal, const float* __restrict__ as2,
                         const float* __restrict__ ad2, float* __restrict__ attn,
                         unsigned* __restrict__ amax) {
    int et = blockIdx.x * blockDim.x + threadIdx.x;
    if (et >= ETOT) return;
    int s, d; float e0, e1;
    if (et < EE) { s = ei[et]; d = ei[EE + et]; e0 = ea[2 * et]; e1 = ea[2 * et + 1]; }
    else         { s = d = et - EE;             e0 = scal[2];    e1 = scal[3]; }
    float a = as2[s] + ad2[d] + e0 * scal[12] + e1 * scal[13];
    a = a > 0.f ? a : 0.2f * a;
    attn[et] = a;
    atomicMax(&amax[d], fmap(a));
}

__global__ void k_ex2(const int* __restrict__ ei, float* __restrict__ attn,
                      const unsigned* __restrict__ amax, float* __restrict__ den) {
    int et = blockIdx.x * blockDim.x + threadIdx.x;
    if (et >= ETOT) return;
    int d = (et < EE) ? ei[EE + et] : et - EE;
    float e = __expf(attn[et] - funmap(amax[d]));
    attn[et] = e;
    atomicAdd(&den[d], e);
}

__global__ void k_agg2(const int* __restrict__ ei, const float* __restrict__ attn,
                       const float* __restrict__ den, const float* __restrict__ h2,
                       float* __restrict__ out2) {
    int eid = (blockIdx.x * blockDim.x + threadIdx.x) >> 5;
    if (eid >= ETOT) return;
    int lane = threadIdx.x & 31;
    int s, d;
    if (eid < EE) { s = ei[eid]; d = ei[EE + eid]; } else { s = d = eid - EE; }
    float w = attn[eid] / (den[d] + 1e-16f);
    const float* hr = h2 + (size_t)s * HID;
    float* orow = out2 + (size_t)d * HID;
    atomicAdd(&orow[lane],      w * hr[lane]);
    atomicAdd(&orow[lane + 32], w * hr[lane + 32]);
}

// final: out[n] = (out2[n]+b2) . Wf + bf  — one wave per node
__global__ void k_final(const float* __restrict__ out2, const float* __restrict__ b2,
                        const float* __restrict__ Wf, const float* __restrict__ bfv,
                        float* __restrict__ out) {
    int wid = (blockIdx.x * blockDim.x + threadIdx.x) >> 5;
    if (wid >= NN) return;
    int lane = threadIdx.x & 31;
    const float* orow = out2 + (size_t)wid * HID;
    float v = (orow[lane] + b2[lane]) * Wf[lane] +
              (orow[lane + 32] + b2[lane + 32]) * Wf[lane + 32];
    for (int o = 16; o; o >>= 1) v += __shfl_xor(v, o);
    if (lane == 0) out[wid] = v + bfv[0];
}

// ---------------------------------------------------------------------------
// Launch
// ---------------------------------------------------------------------------
extern "C" void kernel_launch(void* const* d_in, const int* in_sizes, int n_in,
                              void* d_out, int out_size, void* d_ws, size_t ws_size,
                              hipStream_t stream) {
    const float* x     = (const float*)d_in[0];
    const float* ea    = (const float*)d_in[1];
    const int*   ei    = (const int*)  d_in[2];
    const float* W1    = (const float*)d_in[3];
    const float* asrc1 = (const float*)d_in[4];
    const float* adst1 = (const float*)d_in[5];
    const float* We1   = (const float*)d_in[6];
    const float* atte1 = (const float*)d_in[7];
    const float* b1    = (const float*)d_in[8];
    const float* W2    = (const float*)d_in[9];
    const float* asrc2 = (const float*)d_in[10];
    const float* adst2 = (const float*)d_in[11];
    const float* We2   = (const float*)d_in[12];
    const float* atte2 = (const float*)d_in[13];
    const float* b2    = (const float*)d_in[14];
    const float* Wf    = (const float*)d_in[15];
    const float* bfv   = (const float*)d_in[16];
    float* out = (float*)d_out;

    // workspace carve-up
    char* W = (char*)d_ws;
    size_t off = 0;
    auto carve = [&](size_t bytes) -> char* {
        char* p = W + off;
        off = (off + bytes + 255) & ~(size_t)255;
        return p;
    };
    float*    scal  = (float*)   carve(64);
    bf16_t*   xb    = (bf16_t*)  carve((size_t)NN * KP1 * 2);
    bf16_t*   w1bt  = (bf16_t*)  carve((size_t)C1 * KP1 * 2);
    float*    h1    = (float*)   carve((size_t)NN * C1 * 4);
    float*    as1   = (float*)   carve((size_t)NN * 4 * 4);
    float*    ad1   = (float*)   carve((size_t)NN * 4 * 4);
    unsigned* amax1 = (unsigned*)carve((size_t)NN * 4 * 4);
    float*    den1  = (float*)   carve((size_t)NN * 4 * 4);
    float*    attn1 = (float*)   carve((size_t)ETOT * 4 * 4);
    float*    out1  = (float*)   carve((size_t)NN * C1 * 4);
    bf16_t*   h1b   = (bf16_t*)  carve((size_t)NN * C1 * 2);
    bf16_t*   w2t   = (bf16_t*)  carve((size_t)HID * C1 * 2);
    float*    h2    = (float*)   carve((size_t)NN * HID * 4);
    float*    as2   = (float*)   carve((size_t)NN * 4);
    float*    ad2   = (float*)   carve((size_t)NN * 4);
    unsigned* amax2 = (unsigned*)carve((size_t)NN * 4);
    float*    den2  = (float*)   carve((size_t)NN * 4);
    float*    attn2 = (float*)   carve((size_t)ETOT * 4);
    float*    out2  = (float*)   carve((size_t)NN * HID * 4);
    if (off > ws_size) return;   // workspace too small: bail

    // zero accumulators
    hipMemsetAsync(scal,  0, 64, stream);
    hipMemsetAsync(amax1, 0, (size_t)NN * 4 * 4, stream);
    hipMemsetAsync(den1,  0, (size_t)NN * 4 * 4, stream);
    hipMemsetAsync(out1,  0, (size_t)NN * C1 * 4, stream);
    hipMemsetAsync(amax2, 0, (size_t)NN * 4, stream);
    hipMemsetAsync(den2,  0, (size_t)NN * 4, stream);
    hipMemsetAsync(out2,  0, (size_t)NN * HID * 4, stream);

    const int T = 256;
    auto cdiv = [](long long a, long long b) { return (int)((a + b - 1) / b); };

    // scalars: mean edge_attr + reduced edge-projection vectors
    k_sum_ea<<<cdiv(EE, T), T, 0, stream>>>(ea, scal);
    k_prep<<<1, 32, 0, stream>>>(We1, atte1, We2, atte2, scal);

    // conv1 GEMM
    k_cvt_x  <<<cdiv((long long)NN * KP1, T), T, 0, stream>>>(x, xb);
    k_cvt_w1t<<<cdiv((long long)C1 * KP1, T), T, 0, stream>>>(W1, w1bt);
    k_gemm1  <<<NN / 16, 128, 0, stream>>>(xb, w1bt, h1);

    // conv1 attention + aggregation
    k_att1  <<<cdiv((long long)NN * 32, T), T, 0, stream>>>(h1, asrc1, adst1, as1, ad1);
    k_alpha1<<<cdiv(ETOT, T), T, 0, stream>>>(ei, ea, scal, as1, ad1, attn1, amax1);
    k_ex1   <<<cdiv(ETOT, T), T, 0, stream>>>(ei, attn1, amax1, den1);
    k_agg1  <<<cdiv((long long)ETOT * 32, T), T, 0, stream>>>(ei, attn1, den1, h1, out1);
    k_elu   <<<cdiv((long long)NN * C1, T), T, 0, stream>>>(out1, b1, h1b);

    // conv2 GEMM
    k_cvt_w2t<<<cdiv((long long)HID * C1, T), T, 0, stream>>>(W2, w2t);
    k_gemm2  <<<cdiv(NN / 16, 4), 128, 0, stream>>>(h1b, w2t, h2);

    // conv2 attention + aggregation
    k_att2  <<<cdiv((long long)NN * 32, T), T, 0, stream>>>(h2, asrc2, adst2, as2, ad2);
    k_alpha2<<<cdiv(ETOT, T), T, 0, stream>>>(ei, ea, scal, as2, ad2, attn2, amax2);
    k_ex2   <<<cdiv(ETOT, T), T, 0, stream>>>(ei, attn2, amax2, den2);
    k_agg2  <<<cdiv((long long)ETOT * 32, T), T, 0, stream>>>(ei, attn2, den2, h2, out2);

    // fc head
    k_final <<<cdiv((long long)NN * 32, T), T, 0, stream>>>(out2, b2, Wf, bfv, out);
}
